// FlashAttention_2491081032425
// MI455X (gfx1250) — matrix operand order repaired and checked by reading, not compiled, not run on hardware
//
#include <hip/hip_runtime.h>

// ---- CDNA5 / gfx1250 wave32 WMMA attention ----
typedef __attribute__((ext_vector_type(16))) __bf16 v16bf;
typedef __attribute__((ext_vector_type(8)))  __bf16 v8bf;
typedef __attribute__((ext_vector_type(4)))  __bf16 v4bf;
typedef __attribute__((ext_vector_type(8)))  float  v8f;

union V16 { v16bf v; v8bf h[2]; };

#define D_MODEL 1024
#define SEQ     2048
#define BATCH   4
#define NHEADS  16
#define HDIM    64

static __device__ __forceinline__ v8f wmma_bf16(v16bf a, v16bf b, v8f c) {
  // D = A(16x32 bf16) x B(32x16 bf16) + C(16x16 f32)
  return __builtin_amdgcn_wmma_f32_16x16x32_bf16(false, a, false, b, (short)0, c,
                                                 false, false);
}

static __device__ __forceinline__ v8bf cvt8(float4 a, float4 b) {
  v8bf r = {(__bf16)a.x, (__bf16)a.y, (__bf16)a.z, (__bf16)a.w,
            (__bf16)b.x, (__bf16)b.y, (__bf16)b.z, (__bf16)b.w};
  return r;
}

// ---------------------------------------------------------------------------
// Projection GEMM: Y[m,n] = (sum_k X[m,k] * W[n,k] + bias[n]) * scale  (bf16 out)
// Block: 256 thr (8 waves), tile 64(M) x 128(N); K-chunk 64, LDS double-buffered
// (one barrier per chunk; next chunk's global loads issued before the WMMAs).
// Each wave computes a 32x32 output (2x2 accumulators, 8 WMMAs per chunk).
// ---------------------------------------------------------------------------
__global__ __launch_bounds__(256) void proj_gemm_kernel(
    const float* __restrict__ X, const float* __restrict__ W,
    const float* __restrict__ bias, __bf16* __restrict__ Y, float scale) {
  __shared__ __bf16 As[2][64][64];
  __shared__ __bf16 Bs[2][128][64];
  const int K = D_MODEL, N = D_MODEL;
  const int NSTEP = K / 64;

  const int tid  = threadIdx.x;
  const int lane = tid & 31, wid = tid >> 5;
  const int half = lane >> 4, l16 = lane & 15;
  const int wm = wid >> 2, wn = wid & 3;
  const int m0 = blockIdx.y * 64;
  const int n0 = blockIdx.x * 128;

  // Staging thread -> one 16-float row segment: row = tid>>2, cols (tid&3)*16..+15
  const int sr = tid >> 2;          // 0..63
  const int sc0 = (tid & 3) * 16;   // 0/16/32/48

  v8f acc[2][2] = {};
  float4 ar[4], br[8];

  // ---- prologue: stage chunk 0 into buffer 0 ----
  {
    const float* ap = X + (size_t)(m0 + sr) * K + sc0;
    const float* bp0 = W + (size_t)(n0 + sr) * K + sc0;
    const float* bp1 = W + (size_t)(n0 + sr + 64) * K + sc0;
#pragma unroll
    for (int j = 0; j < 4; ++j) ar[j] = *(const float4*)(ap + j * 4);
#pragma unroll
    for (int j = 0; j < 4; ++j) br[j] = *(const float4*)(bp0 + j * 4);
#pragma unroll
    for (int j = 0; j < 4; ++j) br[4 + j] = *(const float4*)(bp1 + j * 4);
#pragma unroll
    for (int j = 0; j < 2; ++j) {
      *(v8bf*)&As[0][sr][sc0 + j * 8]       = cvt8(ar[2 * j], ar[2 * j + 1]);
      *(v8bf*)&Bs[0][sr][sc0 + j * 8]       = cvt8(br[2 * j], br[2 * j + 1]);
      *(v8bf*)&Bs[0][sr + 64][sc0 + j * 8]  = cvt8(br[4 + 2 * j], br[5 + 2 * j]);
    }
  }
  __syncthreads();

  for (int step = 0; step < NSTEP; ++step) {
    const int cur = step & 1, nxt = cur ^ 1;
    const bool more = (step + 1) < NSTEP;

    // ---- issue next chunk's global loads first (overlap with WMMAs) ----
    if (more) {
      const int kc = (step + 1) * 64;
      const float* ap = X + (size_t)(m0 + sr) * K + kc + sc0;
      const float* bp0 = W + (size_t)(n0 + sr) * K + kc + sc0;
      const float* bp1 = W + (size_t)(n0 + sr + 64) * K + kc + sc0;
#pragma unroll
      for (int j = 0; j < 4; ++j) ar[j] = *(const float4*)(ap + j * 4);
#pragma unroll
      for (int j = 0; j < 4; ++j) br[j] = *(const float4*)(bp0 + j * 4);
#pragma unroll
      for (int j = 0; j < 4; ++j) br[4 + j] = *(const float4*)(bp1 + j * 4);
    }

    // ---- compute on current buffer: 2 sub-K x (2x2) = 8 WMMAs ----
#pragma unroll
    for (int kk = 0; kk < 2; ++kk) {
      V16 af[2], bf[2];
#pragma unroll
      for (int mi = 0; mi < 2; ++mi) {     // A frag: m=l16, k runs half*8 / 16+half*8
        int row = wm * 32 + mi * 16 + l16;
        af[mi].h[0] = *(const v8bf*)&As[cur][row][kk * 32 + half * 8];
        af[mi].h[1] = *(const v8bf*)&As[cur][row][kk * 32 + 16 + half * 8];
      }
#pragma unroll
      for (int ni = 0; ni < 2; ++ni) {     // B frag: n=l16, k runs half*8 / 16+half*8
        int row = wn * 32 + ni * 16 + l16;
        bf[ni].h[0] = *(const v8bf*)&Bs[cur][row][kk * 32 + half * 8];
        bf[ni].h[1] = *(const v8bf*)&Bs[cur][row][kk * 32 + 16 + half * 8];
      }
#pragma unroll
      for (int mi = 0; mi < 2; ++mi)
#pragma unroll
        for (int ni = 0; ni < 2; ++ni)
          acc[mi][ni] = wmma_bf16(af[mi].v, bf[ni].v, acc[mi][ni]);
    }

    // ---- store next chunk into the other buffer ----
    if (more) {
#pragma unroll
      for (int j = 0; j < 2; ++j) {
        *(v8bf*)&As[nxt][sr][sc0 + j * 8]      = cvt8(ar[2 * j], ar[2 * j + 1]);
        *(v8bf*)&Bs[nxt][sr][sc0 + j * 8]      = cvt8(br[2 * j], br[2 * j + 1]);
        *(v8bf*)&Bs[nxt][sr + 64][sc0 + j * 8] = cvt8(br[4 + 2 * j], br[5 + 2 * j]);
      }
    }
    __syncthreads();
  }

#pragma unroll
  for (int ni = 0; ni < 2; ++ni) {
    int n = n0 + wn * 32 + ni * 16 + l16;
    float bv = bias[n];
#pragma unroll
    for (int mi = 0; mi < 2; ++mi)
#pragma unroll
      for (int r = 0; r < 8; ++r) {        // C row = r + half*8
        int m = m0 + wm * 32 + mi * 16 + r + half * 8;
        Y[(size_t)m * N + n] = (__bf16)((acc[mi][ni][r] + bv) * scale);
      }
  }
}

// ---------------------------------------------------------------------------
// Output GEMM: out[m,n] = sum_k A[m,k](bf16) * W[n,k](f32) + bias[n]  (f32 out)
// Same double-buffered structure; A is already bf16.
// ---------------------------------------------------------------------------
__global__ __launch_bounds__(256) void out_gemm_kernel(
    const __bf16* __restrict__ A, const float* __restrict__ W,
    const float* __restrict__ bias, float* __restrict__ out) {
  __shared__ __bf16 As[2][64][64];
  __shared__ __bf16 Bs[2][128][64];
  const int K = D_MODEL, N = D_MODEL;
  const int NSTEP = K / 64;

  const int tid  = threadIdx.x;
  const int lane = tid & 31, wid = tid >> 5;
  const int half = lane >> 4, l16 = lane & 15;
  const int wm = wid >> 2, wn = wid & 3;
  const int m0 = blockIdx.y * 64;
  const int n0 = blockIdx.x * 128;

  const int sr = tid >> 2;
  const int sc0 = (tid & 3) * 16;

  v8f acc[2][2] = {};
  v8bf arb[2];
  float4 br[8];

  {
    const __bf16* ap = A + (size_t)(m0 + sr) * K + sc0;
    const float* bp0 = W + (size_t)(n0 + sr) * K + sc0;
    const float* bp1 = W + (size_t)(n0 + sr + 64) * K + sc0;
    arb[0] = *(const v8bf*)(ap);
    arb[1] = *(const v8bf*)(ap + 8);
#pragma unroll
    for (int j = 0; j < 4; ++j) br[j] = *(const float4*)(bp0 + j * 4);
#pragma unroll
    for (int j = 0; j < 4; ++j) br[4 + j] = *(const float4*)(bp1 + j * 4);
    *(v8bf*)&As[0][sr][sc0]     = arb[0];
    *(v8bf*)&As[0][sr][sc0 + 8] = arb[1];
#pragma unroll
    for (int j = 0; j < 2; ++j) {
      *(v8bf*)&Bs[0][sr][sc0 + j * 8]      = cvt8(br[2 * j], br[2 * j + 1]);
      *(v8bf*)&Bs[0][sr + 64][sc0 + j * 8] = cvt8(br[4 + 2 * j], br[5 + 2 * j]);
    }
  }
  __syncthreads();

  for (int step = 0; step < NSTEP; ++step) {
    const int cur = step & 1, nxt = cur ^ 1;
    const bool more = (step + 1) < NSTEP;

    if (more) {
      const int kc = (step + 1) * 64;
      const __bf16* ap = A + (size_t)(m0 + sr) * K + kc + sc0;
      const float* bp0 = W + (size_t)(n0 + sr) * K + kc + sc0;
      const float* bp1 = W + (size_t)(n0 + sr + 64) * K + kc + sc0;
      arb[0] = *(const v8bf*)(ap);
      arb[1] = *(const v8bf*)(ap + 8);
#pragma unroll
      for (int j = 0; j < 4; ++j) br[j] = *(const float4*)(bp0 + j * 4);
#pragma unroll
      for (int j = 0; j < 4; ++j) br[4 + j] = *(const float4*)(bp1 + j * 4);
    }

#pragma unroll
    for (int kk = 0; kk < 2; ++kk) {
      V16 af[2], bf[2];
#pragma unroll
      for (int mi = 0; mi < 2; ++mi) {
        int row = wm * 32 + mi * 16 + l16;
        af[mi].h[0] = *(const v8bf*)&As[cur][row][kk * 32 + half * 8];
        af[mi].h[1] = *(const v8bf*)&As[cur][row][kk * 32 + 16 + half * 8];
      }
#pragma unroll
      for (int ni = 0; ni < 2; ++ni) {
        int row = wn * 32 + ni * 16 + l16;
        bf[ni].h[0] = *(const v8bf*)&Bs[cur][row][kk * 32 + half * 8];
        bf[ni].h[1] = *(const v8bf*)&Bs[cur][row][kk * 32 + 16 + half * 8];
      }
#pragma unroll
      for (int mi = 0; mi < 2; ++mi)
#pragma unroll
        for (int ni = 0; ni < 2; ++ni)
          acc[mi][ni] = wmma_bf16(af[mi].v, bf[ni].v, acc[mi][ni]);
    }

    if (more) {
      *(v8bf*)&As[nxt][sr][sc0]     = arb[0];
      *(v8bf*)&As[nxt][sr][sc0 + 8] = arb[1];
#pragma unroll
      for (int j = 0; j < 2; ++j) {
        *(v8bf*)&Bs[nxt][sr][sc0 + j * 8]      = cvt8(br[2 * j], br[2 * j + 1]);
        *(v8bf*)&Bs[nxt][sr + 64][sc0 + j * 8] = cvt8(br[4 + 2 * j], br[5 + 2 * j]);
      }
    }
    __syncthreads();
  }

#pragma unroll
  for (int ni = 0; ni < 2; ++ni) {
    int n = n0 + wn * 32 + ni * 16 + l16;
    float bv = bias[n];
#pragma unroll
    for (int mi = 0; mi < 2; ++mi)
#pragma unroll
      for (int r = 0; r < 8; ++r) {
        int m = m0 + wm * 32 + mi * 16 + r + half * 8;
        out[(size_t)m * N + n] = acc[mi][ni][r] + bv;
      }
  }
}

// ---------------------------------------------------------------------------
// Flash attention. Qh already includes the 1/sqrt(Dh) scale.
// Layouts: Qh/Kh/Vh/Oc are [B, S, H*Dh] bf16 (head h at column h*64).
// Block: 128 thr (4 waves); each wave owns 16 query rows; block = 64 rows.
// K/V tiles of 32 keys, LDS double-buffered (1 barrier/tile, loads ahead of
// WMMAs). Row sums via an extra WMMA against an all-ones B matrix; row max
// via a single 4-step xor-shuffle reduce of the tile-wise elementwise max.
// ---------------------------------------------------------------------------
__global__ __launch_bounds__(128) void attn_kernel(
    const __bf16* __restrict__ Qh, const __bf16* __restrict__ Kh,
    const __bf16* __restrict__ Vh, __bf16* __restrict__ Oc) {
  __shared__ __bf16 Kt[2][32][64];     // [buf][key][d]
  __shared__ __bf16 Vt[2][64][32];     // [buf][d][key]  (transposed)
  __shared__ __bf16 Pl[4][16][32];     // per-wave P re-layout scratch

  const int D = D_MODEL;
  const int NT = SEQ / 32;
  const int tid  = threadIdx.x;
  const int lane = tid & 31, wid = tid >> 5;
  const int half = lane >> 4, l16 = lane & 15;

  const int b = blockIdx.z, h = blockIdx.y;
  const int s0 = blockIdx.x * 64 + wid * 16;
  const size_t base = (size_t)b * SEQ * D + (size_t)h * HDIM;

  // Q A-fragments (row = l16; K runs per ISA A layout), reused for all tiles.
  v16bf qa[2];
  {
    const size_t rowoff = base + (size_t)(s0 + l16) * D;
#pragma unroll
    for (int kk = 0; kk < 2; ++kk) {
      V16 t;
      t.h[0] = *(const v8bf*)(Qh + rowoff + kk * 32 + half * 8);
      t.h[1] = *(const v8bf*)(Qh + rowoff + kk * 32 + 16 + half * 8);
      qa[kk] = t.v;
    }
  }

  v16bf onesv;
#pragma unroll
  for (int i = 0; i < 16; ++i) onesv[i] = (__bf16)1.0f;

  v8f acc[4] = {};
  v8f mrow, lrow;
#pragma unroll
  for (int i = 0; i < 8; ++i) { mrow[i] = -1e30f; lrow[i] = 0.0f; }

  const int kt_t  = tid >> 2;          // 0..31: key row this thread stages
  const int kt_d0 = (tid & 3) * 16;    // 0/16/32/48

  // ---- prologue: stage tile 0 into buffer 0 ----
  {
    const size_t roff = base + (size_t)kt_t * D + kt_d0;
    v8bf k0 = *(const v8bf*)(Kh + roff);
    v8bf k1 = *(const v8bf*)(Kh + roff + 8);
    v8bf v0 = *(const v8bf*)(Vh + roff);
    v8bf v1 = *(const v8bf*)(Vh + roff + 8);
    *(v8bf*)&Kt[0][kt_t][kt_d0]     = k0;
    *(v8bf*)&Kt[0][kt_t][kt_d0 + 8] = k1;
#pragma unroll
    for (int j = 0; j < 8; ++j) {
      Vt[0][kt_d0 + j][kt_t]     = v0[j];
      Vt[0][kt_d0 + 8 + j][kt_t] = v1[j];
    }
  }
  __syncthreads();

  for (int it = 0; it < NT; ++it) {
    const int cur = it & 1, nxt = cur ^ 1;
    const bool more = (it + 1) < NT;

    // ---- issue next tile's global loads (overlap with this tile's math) ----
    v8bf k0, k1, v0, v1;
    if (more) {
      const size_t roff = base + (size_t)((it + 1) * 32 + kt_t) * D + kt_d0;
      k0 = *(const v8bf*)(Kh + roff);
      k1 = *(const v8bf*)(Kh + roff + 8);
      v0 = *(const v8bf*)(Vh + roff);
      v1 = *(const v8bf*)(Vh + roff + 8);
    }

    // ---- scores: two 16x16 tiles, K-dim = Dh = 64 ----
    v8f sc[2];
#pragma unroll
    for (int nt = 0; nt < 2; ++nt) {
      v8f s = {};
#pragma unroll
      for (int kk = 0; kk < 2; ++kk) {
        V16 kb;
        kb.h[0] = *(const v8bf*)&Kt[cur][nt * 16 + l16][kk * 32 + half * 8];
        kb.h[1] = *(const v8bf*)&Kt[cur][nt * 16 + l16][kk * 32 + 16 + half * 8];
        s = wmma_bf16(qa[kk], kb.v, s);
      }
      sc[nt] = s;
    }

    // ---- online softmax: max via xor-shuffles on elementwise tile max ----
    v8f rmx;
#pragma unroll
    for (int i = 0; i < 8; ++i) rmx[i] = fmaxf(sc[0][i], sc[1][i]);
#pragma unroll
    for (int msk = 1; msk <= 8; msk <<= 1)
#pragma unroll
      for (int i = 0; i < 8; ++i)
        rmx[i] = fmaxf(rmx[i], __shfl_xor(rmx[i], msk, 32));

    v8f mnew, corr, p0, p1;
#pragma unroll
    for (int i = 0; i < 8; ++i) {
      mnew[i] = fmaxf(mrow[i], rmx[i]);
      corr[i] = __expf(mrow[i] - mnew[i]);
      p0[i]   = __expf(sc[0][i] - mnew[i]);
      p1[i]   = __expf(sc[1][i] - mnew[i]);
      mrow[i] = mnew[i];
    }
#pragma unroll
    for (int nt = 0; nt < 4; ++nt)
      acc[nt] *= corr;

    // ---- re-layout P (C layout -> A layout) through per-wave LDS ----
#pragma unroll
    for (int r = 0; r < 8; ++r) {
      Pl[wid][r + half * 8][l16]      = (__bf16)p0[r];
      Pl[wid][r + half * 8][16 + l16] = (__bf16)p1[r];
    }
    asm volatile("" ::: "memory");   // keep ds store->load order (same-wave LDS is in-order)
    V16 pf;
    pf.h[0] = *(const v8bf*)&Pl[wid][l16][half * 8];
    pf.h[1] = *(const v8bf*)&Pl[wid][l16][16 + half * 8];

    // ---- row sums: one WMMA against all-ones B (C rows align with lrow) ----
    {
      v8f rs = {};
      rs = wmma_bf16(pf.v, onesv, rs);
#pragma unroll
      for (int i = 0; i < 8; ++i)
        lrow[i] = lrow[i] * corr[i] + rs[i];
    }

    // ---- O += P x V : 4 d-tiles, contiguous B-frags from transposed Vt ----
#pragma unroll
    for (int nt = 0; nt < 4; ++nt) {
      V16 vb;
      vb.h[0] = *(const v8bf*)&Vt[cur][nt * 16 + l16][half * 8];
      vb.h[1] = *(const v8bf*)&Vt[cur][nt * 16 + l16][16 + half * 8];
      acc[nt] = wmma_bf16(pf.v, vb.v, acc[nt]);
    }

    // ---- stage next tile into the other buffer ----
    if (more) {
      *(v8bf*)&Kt[nxt][kt_t][kt_d0]     = k0;
      *(v8bf*)&Kt[nxt][kt_t][kt_d0 + 8] = k1;
#pragma unroll
      for (int j = 0; j < 8; ++j) {
        Vt[nxt][kt_d0 + j][kt_t]     = v0[j];
        Vt[nxt][kt_d0 + 8 + j][kt_t] = v1[j];
      }
    }
    __syncthreads();
  }

  // ---- normalize and write O (bf16, [B,S,H*Dh]) ----
#pragma unroll
  for (int nt = 0; nt < 4; ++nt) {
    int d = nt * 16 + l16;
#pragma unroll
    for (int r = 0; r < 8; ++r) {
      int m = s0 + r + half * 8;
      float o = acc[nt][r] / lrow[r];
      Oc[base + (size_t)m * D + d] = (__bf16)o;
    }
  }
}

// ---------------------------------------------------------------------------
extern "C" void kernel_launch(void* const* d_in, const int* in_sizes, int n_in,
                              void* d_out, int out_size, void* d_ws, size_t ws_size,
                              hipStream_t stream) {
  const float* q  = (const float*)d_in[0];
  const float* k  = (const float*)d_in[1];
  const float* v  = (const float*)d_in[2];
  const float* Wq = (const float*)d_in[3];
  const float* bq = (const float*)d_in[4];
  const float* Wk = (const float*)d_in[5];
  const float* bk = (const float*)d_in[6];
  const float* Wv = (const float*)d_in[7];
  const float* bv = (const float*)d_in[8];
  const float* Wo = (const float*)d_in[9];
  const float* bo = (const float*)d_in[10];
  float* out = (float*)d_out;

  const size_t NE = (size_t)BATCH * SEQ * D_MODEL;   // 8M elements
  char* ws = (char*)d_ws;
  __bf16* Qh = (__bf16*)(ws);
  __bf16* Kh = (__bf16*)(ws + 2 * NE);
  __bf16* Vh = (__bf16*)(ws + 4 * NE);
  __bf16* Oc = (__bf16*)(ws + 6 * NE);

  dim3 gg(D_MODEL / 128, (BATCH * SEQ) / 64);        // (8, 128)
  dim3 gb(256);
  // 1/sqrt(Dh) folded into the Q projection (weights AND bias).
  proj_gemm_kernel<<<gg, gb, 0, stream>>>(q, Wq, bq, Qh, 0.125f);
  proj_gemm_kernel<<<gg, gb, 0, stream>>>(k, Wk, bk, Kh, 1.0f);
  proj_gemm_kernel<<<gg, gb, 0, stream>>>(v, Wv, bv, Vh, 1.0f);

  attn_kernel<<<dim3(SEQ / 64, NHEADS, BATCH), 128, 0, stream>>>(Qh, Kh, Vh, Oc);

  out_gemm_kernel<<<gg, gb, 0, stream>>>(Oc, Wo, bo, out);
}
